// ThinStackHybridLSTM_2929167696067
// MI455X (gfx1250) — compile-verified, matrix operand order checked by reading, not run on hardware
//
#include <hip/hip_runtime.h>
#include <math.h>

// Dims: V=100000 E=300 H=256 TK=64 O=5 B=128 L=64 T=123
typedef __attribute__((ext_vector_type(16))) _Float16 v16h;
typedef __attribute__((ext_vector_type(8)))  float    v8f;

__device__ __forceinline__ int kmap(int e, int half) {
  // 16-bit A/B fragment: element e -> K offset within a 32-wide chunk
  // e=0..7 -> k = e + 8*half ; e=8..15 -> k = 16 + (e-8) + 8*half
  return (e & 7) + ((e >> 3) << 4) + (half << 3);
}

// Contiguous f16 fragment: two 16-byte runs per fragment -> vector loads
__device__ __forceinline__ v16h frag_h(const _Float16* __restrict__ row,
                                       int kbase, int half) {
  v16h f;
#pragma unroll
  for (int e = 0; e < 16; ++e) f[e] = row[kbase + kmap(e, half)];
  return f;
}

__device__ __forceinline__ float sigf(float x) { return 1.0f / (1.0f + expf(-x)); }

// ---------------------------------------------------------------------------
// One-time staging: all WMMA operands to padded f16.
//   Wg   = [W_ih | W_hh]           256 x 832
//   Wc   = [Wl | Wr | Wt]          1280 x 576
//   Winh = W_in zero-padded        256 x 320   (K=300 -> 320)
//   xemb = embed[tokens] padded    8192 x 320
// ---------------------------------------------------------------------------
__global__ void stage_kernel(const int*   __restrict__ tokens,
                             const float* __restrict__ embed,
                             const float* __restrict__ Winw,
                             const float* __restrict__ Wih, const float* __restrict__ Whh,
                             const float* __restrict__ Wl,  const float* __restrict__ Wr,
                             const float* __restrict__ Wt,
                             _Float16* __restrict__ Wg,   _Float16* __restrict__ Wc,
                             _Float16* __restrict__ Winh, _Float16* __restrict__ xemb) {
  const int N1 = 256 * 832;            // Wg
  const int N2 = N1 + 1280 * 576;      // Wc
  const int N3 = N2 + 256 * 320;       // Winh
  const int N4 = N3 + 8192 * 320;      // xemb
  for (int i = blockIdx.x * blockDim.x + threadIdx.x; i < N4;
       i += gridDim.x * blockDim.x) {
    if (i < N1) {
      int n = i / 832, k = i % 832;
      float v = (k < 768) ? Wih[n * 768 + k] : Whh[n * 64 + (k - 768)];
      Wg[i] = (_Float16)v;
    } else if (i < N2) {
      int j = i - N1;
      int n = j / 576, k = j % 576;
      float v = (k < 256) ? Wl[n * 256 + k]
              : (k < 512) ? Wr[n * 256 + (k - 256)]
                          : Wt[n * 64 + (k - 512)];
      Wc[j] = (_Float16)v;
    } else if (i < N3) {
      int j = i - N2;
      int n = j / 320, k = j % 320;
      Winh[j] = (k < 300) ? (_Float16)Winw[n * 300 + k] : (_Float16)0.0f;
    } else {
      int j = i - N3;
      int m = j / 320, k = j % 320;
      int tok = tokens[m];
      xemb[j] = (k < 300) ? (_Float16)embed[(long)tok * 300 + k] : (_Float16)0.0f;
    }
  }
}

// ---------------------------------------------------------------------------
// Phase 0: x = selu(xemb @ Winh^T + b), stored reversed as buf[b][L-1-l]
// 8192 waves, one 16x16 tile each; K = 320 (10 chunks), padding is zeros.
// ---------------------------------------------------------------------------
__global__ void phase0_embed(const _Float16* __restrict__ xemb,
                             const _Float16* __restrict__ Winh,
                             const float* __restrict__ Winb,
                             float* __restrict__ buf) {
  int wave = (blockIdx.x * blockDim.x + threadIdx.x) >> 5;
  int lane = threadIdx.x & 31;
  int half = lane >> 4;
  int r16  = lane & 15;
  int mt = wave >> 4;      // 0..511
  int nt = wave & 15;      // 0..15
  if (mt >= 512) return;

  const _Float16* arow = xemb + (mt * 16 + r16) * 320;
  const _Float16* brow = Winh + (nt * 16 + r16) * 320;

  v8f acc = {};
  for (int kc = 0; kc < 10; ++kc) {
    v16h a = frag_h(arow, kc * 32, half);
    v16h b = frag_h(brow, kc * 32, half);
    acc = __builtin_amdgcn_wmma_f32_16x16x32_f16(false, a, false, b,
                                                 (short)0, acc, false, false);
  }
  int n = nt * 16 + r16;
#pragma unroll
  for (int r = 0; r < 8; ++r) {
    int mm = mt * 16 + r + 8 * half;
    int b = mm >> 6, l = mm & 63;
    float v = acc[r] + Winb[n];
    float s = (v > 0.f) ? 1.0507009873554805f * v
                        : 1.0507009873554805f * 1.6732632423543772f * (expf(v) - 1.f);
    buf[((b * 64) + (63 - l)) * 256 + n] = s;   // reversed along l
  }
}

// ---------------------------------------------------------------------------
// out0[l][b][o] = log_softmax(x[b][l] @ W_out^T + b_out)
// ---------------------------------------------------------------------------
__global__ void out0_kernel(const float* __restrict__ buf,
                            const float* __restrict__ Wout,
                            const float* __restrict__ Woutb,
                            float* __restrict__ out) {
  int gid = blockIdx.x * blockDim.x + threadIdx.x;
  if (gid >= 8192) return;
  int b = gid >> 6, l = gid & 63;
  const float* xr = buf + ((b * 64) + (63 - l)) * 256;   // un-reverse
  float lg[5];
  for (int o = 0; o < 5; ++o) {
    float s = Woutb[o];
    for (int h = 0; h < 256; ++h) s += xr[h] * Wout[o * 256 + h];
    lg[o] = s;
  }
  float mx = lg[0];
  for (int o = 1; o < 5; ++o) mx = fmaxf(mx, lg[o]);
  float sm = 0.f;
  for (int o = 0; o < 5; ++o) sm += expf(lg[o] - mx);
  float lse = logf(sm);
  for (int o = 0; o < 5; ++o) out[(l * 128 + b) * 5 + o] = lg[o] - mx - lse;
}

// ---------------------------------------------------------------------------
// Persistent sequential kernel: one workgroup, 32 wave32 waves, 123 steps.
// Per-step GEMMs in v_wmma_f32_16x16x32_f16, A staged to f16 between barriers.
// Output tiles processed in pairs (two f32 accumulators) so each A fragment
// is loaded once per K-chunk per pair.
// ---------------------------------------------------------------------------
__global__ void __launch_bounds__(1024) seq_kernel(
    const int*   __restrict__ trans,
    const float* __restrict__ bih,  const float* __restrict__ bhh,
    const float* __restrict__ Wlb,
    const float* __restrict__ Wout, const float* __restrict__ Woutb,
    const _Float16* __restrict__ Wg, const _Float16* __restrict__ Wc,
    const float* __restrict__ buf,
    float* __restrict__ stack, float* __restrict__ th, float* __restrict__ tc,
    float* __restrict__ gates, float* __restrict__ li,
    _Float16* __restrict__ xt, _Float16* __restrict__ xr,
    float* __restrict__ out1) {
  const int tid  = threadIdx.x;
  const int lane = tid & 31;
  const int wv   = tid >> 5;
  const int half = lane >> 4;
  const int r16  = lane & 15;

  // init stack rows 0,1 with pad = buf[0][0] (= x[batch0, L-1]) duplicated (h,c)
  for (int i = tid; i < 128 * 1024; i += 1024) {
    int b = i >> 10, rj = i & 1023;
    int row = rj >> 9, col = rj & 511;
    stack[(b * 66 + row) * 512 + col] = buf[col & 255];
  }
  for (int i = tid; i < 8192; i += 1024) { th[i] = 0.f; tc[i] = 0.f; }
  __syncthreads();

  int sp = 2, bp = 64, oidx = 0;
  for (int t = 0; t < 123; ++t) {
    int tr = trans[t];   // first batch row governs all (matches reference)

    // ---- pack xt = [buf[bp-1] | stack[sp-1].h | stack[sp-2].h | th] as f16
    for (int i = tid; i < 128 * 832; i += 1024) {
      int b = i / 832, k = i % 832;
      float v;
      if (k < 256)      v = buf[(b * 64 + bp - 1) * 256 + k];
      else if (k < 512) v = stack[(b * 66 + sp - 1) * 512 + (k - 256)];
      else if (k < 768) v = stack[(b * 66 + sp - 2) * 512 + (k - 512)];
      else              v = th[b * 64 + (k - 768)];
      xt[i] = (_Float16)v;
    }
    __syncthreads();

    // ---- gates = xt @ [W_ih|W_hh]^T : 128x256 (8x16 tiles), K=832 (26 chunks)
    {
      int mt  = wv >> 2;            // 4 tiles per wave, all in one M-row
      int ntb = (wv & 3) * 4;
      const _Float16* ar = xt + (mt * 16 + r16) * 832;
#pragma unroll
      for (int g = 0; g < 2; ++g) {
        int nt0 = ntb + g * 2;
        const _Float16* b0r = Wg + ((nt0 + 0) * 16 + r16) * 832;
        const _Float16* b1r = Wg + ((nt0 + 1) * 16 + r16) * 832;
        v8f acc0 = {}, acc1 = {};
        for (int kc = 0; kc < 26; ++kc) {
          v16h a  = frag_h(ar,  kc * 32, half);
          v16h b0 = frag_h(b0r, kc * 32, half);
          acc0 = __builtin_amdgcn_wmma_f32_16x16x32_f16(false, a, false, b0,
                                                        (short)0, acc0, false, false);
          v16h b1 = frag_h(b1r, kc * 32, half);
          acc1 = __builtin_amdgcn_wmma_f32_16x16x32_f16(false, a, false, b1,
                                                        (short)0, acc1, false, false);
        }
#pragma unroll
        for (int r = 0; r < 8; ++r) {
          int m = mt * 16 + r + 8 * half;
          gates[m * 256 + (nt0 + 0) * 16 + r16] = acc0[r];
          gates[m * 256 + (nt0 + 1) * 16 + r16] = acc1[r];
        }
      }
    }
    __syncthreads();

    // ---- tracker LSTM cell update
    for (int i = tid; i < 8192; i += 1024) {
      int b = i >> 6, j = i & 63;
      const float* g = gates + b * 256;
      float gi = g[j]       + bih[j]       + bhh[j];
      float gf = g[64 + j]  + bih[64 + j]  + bhh[64 + j];
      float gg = g[128 + j] + bih[128 + j] + bhh[128 + j];
      float go = g[192 + j] + bih[192 + j] + bhh[192 + j];
      float c = sigf(gf) * tc[i] + sigf(gi) * tanhf(gg);
      tc[i] = c;
      th[i] = sigf(go) * tanhf(c);
    }
    __syncthreads();

    if (tr == 1) {
      // ---- shift: push leaf (h=c=buf[bp-1])
      for (int i = tid; i < 128 * 512; i += 1024) {
        int b = i >> 9, j = i & 511;
        stack[(b * 66 + sp) * 512 + j] = buf[(b * 64 + bp - 1) * 256 + (j & 255)];
      }
      sp += 1; bp -= 1;
    } else {
      // ---- pack xr = [left.h | right.h | th] as f16 (128 x 576)
      for (int i = tid; i < 128 * 576; i += 1024) {
        int b = i / 576, k = i % 576;
        float v = (k < 256) ? stack[(b * 66 + sp - 2) * 512 + k]
                : (k < 512) ? stack[(b * 66 + sp - 1) * 512 + (k - 256)]
                            : th[b * 64 + (k - 512)];
        xr[i] = (_Float16)v;
      }
      __syncthreads();

      // ---- lstm_in = xr @ [Wl|Wr|Wt]^T : 128x1280 (8x80 tiles), K=576 (18 chunks)
      {
        int mt  = wv >> 2;            // 20 tiles per wave, all in one M-row
        int ntb = (wv & 3) * 20;
        const _Float16* ar = xr + (mt * 16 + r16) * 576;
        for (int g = 0; g < 10; ++g) {
          int nt0 = ntb + g * 2;
          const _Float16* b0r = Wc + ((nt0 + 0) * 16 + r16) * 576;
          const _Float16* b1r = Wc + ((nt0 + 1) * 16 + r16) * 576;
          v8f acc0 = {}, acc1 = {};
          for (int kc = 0; kc < 18; ++kc) {
            v16h a  = frag_h(ar,  kc * 32, half);
            v16h b0 = frag_h(b0r, kc * 32, half);
            acc0 = __builtin_amdgcn_wmma_f32_16x16x32_f16(false, a, false, b0,
                                                          (short)0, acc0, false, false);
            v16h b1 = frag_h(b1r, kc * 32, half);
            acc1 = __builtin_amdgcn_wmma_f32_16x16x32_f16(false, a, false, b1,
                                                          (short)0, acc1, false, false);
          }
#pragma unroll
          for (int r = 0; r < 8; ++r) {
            int m = mt * 16 + r + 8 * half;
            li[m * 1280 + (nt0 + 0) * 16 + r16] = acc0[r];
            li[m * 1280 + (nt0 + 1) * 16 + r16] = acc1[r];
          }
        }
      }
      __syncthreads();

      // ---- tree-LSTM combine, write [h|c] into stack[sp-2]
      for (int i = tid; i < 128 * 256; i += 1024) {
        int b = i >> 8, j = i & 255;
        const float* L = li + b * 1280;
        float av = L[j]        + Wlb[j];
        float iv = L[256 + j]  + Wlb[256 + j];
        float f1 = L[512 + j]  + Wlb[512 + j];
        float f2 = L[768 + j]  + Wlb[768 + j];
        float ov = L[1024 + j] + Wlb[1024 + j];
        float lc = stack[(b * 66 + sp - 2) * 512 + 256 + j];
        float rc = stack[(b * 66 + sp - 1) * 512 + 256 + j];
        float c = tanhf(av) * sigf(iv) + sigf(f1) * lc + sigf(f2) * rc;
        float h = sigf(ov) * tanhf(c);
        stack[(b * 66 + sp - 2) * 512 + j] = h;
        stack[(b * 66 + sp - 2) * 512 + 256 + j] = c;
      }
      __syncthreads();

      // ---- output head on h
      if (tid < 128) {
        int b = tid;
        const float* hr = stack + (b * 66 + sp - 2) * 512;
        float lg[5];
        for (int o = 0; o < 5; ++o) {
          float s = Woutb[o];
          for (int h = 0; h < 256; ++h) s += hr[h] * Wout[o * 256 + h];
          lg[o] = s;
        }
        float mx = lg[0];
        for (int o = 1; o < 5; ++o) mx = fmaxf(mx, lg[o]);
        float sm = 0.f;
        for (int o = 0; o < 5; ++o) sm += expf(lg[o] - mx);
        float lse = logf(sm);
        for (int o = 0; o < 5; ++o) out1[(oidx * 128 + b) * 5 + o] = lg[o] - mx - lse;
      }
      sp -= 1; oidx += 1;
    }
    __syncthreads();
  }
}

extern "C" void kernel_launch(void* const* d_in, const int* in_sizes, int n_in,
                              void* d_out, int out_size, void* d_ws, size_t ws_size,
                              hipStream_t stream) {
  const int*   tokens = (const int*)d_in[0];
  const int*   trans  = (const int*)d_in[1];
  const float* embed  = (const float*)d_in[2];
  const float* Winw   = (const float*)d_in[3];
  const float* Winb   = (const float*)d_in[4];
  const float* Wlw    = (const float*)d_in[5];
  const float* Wlb    = (const float*)d_in[6];
  const float* Wrw    = (const float*)d_in[7];
  const float* Wtw    = (const float*)d_in[8];
  const float* Wih    = (const float*)d_in[9];
  const float* Whh    = (const float*)d_in[10];
  const float* bih    = (const float*)d_in[11];
  const float* bhh    = (const float*)d_in[12];
  const float* Woutw  = (const float*)d_in[13];
  const float* Woutb  = (const float*)d_in[14];
  float* out = (float*)d_out;

  // workspace carve-up (~34.2 MB), all sub-buffers 16B aligned
  float* buf   = (float*)d_ws;               // 128*64*256 = 2,097,152 f32
  float* stack = buf + 2097152;              // 128*66*512 = 4,325,376 f32
  float* th    = stack + 4325376;            // 8192 f32
  float* tc    = th + 8192;                  // 8192 f32
  float* gates = tc + 8192;                  // 32768 f32
  float* li    = gates + 32768;              // 163840 f32
  _Float16* xt   = (_Float16*)(li + 163840); // 128*832  = 106,496 f16
  _Float16* Wg   = xt + 106496;              // 256*832  = 212,992 f16
  _Float16* Wc   = Wg + 212992;              // 1280*576 = 737,280 f16
  _Float16* xr   = Wc + 737280;              // 128*576  =  73,728 f16
  _Float16* Winh = xr + 73728;               // 256*320  =  81,920 f16
  _Float16* xemb = Winh + 81920;             // 8192*320 = 2,621,440 f16

  stage_kernel<<<2048, 256, 0, stream>>>(tokens, embed, Winw, Wih, Whh,
                                         Wlw, Wrw, Wtw, Wg, Wc, Winh, xemb);
  phase0_embed<<<1024, 256, 0, stream>>>(xemb, Winh, Winb, buf);
  out0_kernel <<<32, 256, 0, stream>>>(buf, Woutw, Woutb, out);
  seq_kernel  <<<1, 1024, 0, stream>>>(trans, bih, bhh, Wlb, Woutw, Woutb,
                                       Wg, Wc, buf, stack, th, tc,
                                       gates, li, xt, xr, out + 64 * 128 * 5);
}
